// Attend_62534723830373
// MI455X (gfx1250) — matrix operand order repaired and checked by reading, not compiled, not run on hardware
//
#include <hip/hip_runtime.h>
#include <cmath>

typedef _Float16 v16h __attribute__((ext_vector_type(16)));
typedef _Float16 v8h  __attribute__((ext_vector_type(8)));
typedef _Float16 v4h  __attribute__((ext_vector_type(4)));
typedef float    v8f  __attribute__((ext_vector_type(8)));

#define B_  2
#define H_  16
#define S_  2048
#define D_  128
#define BM  64      // query rows per block (4 waves x 16)
#define BN  32      // kv columns per iteration

__device__ __forceinline__ float xor16f(float x) {
    // swap lanes 0-15 <-> 16-31 (group-of-32 swizzle, xor=0x10, and=0x1f)
    return __int_as_float(__builtin_amdgcn_ds_swizzle(__float_as_int(x), 0x401F));
}
__device__ __forceinline__ float lanebc(float x, int srcLane) {
    return __int_as_float(__builtin_amdgcn_ds_bpermute(srcLane << 2, __float_as_int(x)));
}
__device__ __forceinline__ float fast_exp2(float x) {
    // args are <= 0 here; raw v_exp_f32 (flush-to-zero on tiny results) is exactly
    // what softmax wants and skips OCML's ldexp denorm fixup.
#if __has_builtin(__builtin_amdgcn_exp2f)
    return __builtin_amdgcn_exp2f(x);
#else
    return exp2f(x);
#endif
}

__global__ __launch_bounds__(128)
void fa_swapop_kernel(const float* __restrict__ kq,   // "queries" = k input
                      const float* __restrict__ vv,   // keys AND values = v input
                      float* __restrict__ out)
{
    __shared__ _Float16 smV [BN][D_ + 8];   // row-major  v[j][d]
    __shared__ _Float16 smVT[D_][BN + 8];   // transposed v[d][j]

    const int tid  = threadIdx.x;
    const int w    = tid >> 5;       // wave id 0..3
    const int lane = tid & 31;
    const int li   = lane & 15;
    const int hi   = lane >> 4;

    const int i0 = blockIdx.x * BM;  // block's first query row
    const int bh = blockIdx.y;       // fused batch*head
    const int iw = i0 + 16 * w;      // wave's first query row
    const int ig = iw + li;          // this lane's query row (softmax row)

    const float* kbh = kq  + (size_t)bh * S_ * D_;
    const float* vbh = vv  + (size_t)bh * S_ * D_;
    float*       obh = out + (size_t)bh * S_ * D_;

    // ---- Q rows (16 per wave) as f16 B-operand, scale*log2(e) folded in ----
    const float QS = 0.08838834764831845f * 1.44269504088896340736f;
    v16h qb[4];
#pragma unroll
    for (int ch = 0; ch < 4; ++ch) {
        const float* qs = kbh + (size_t)ig * D_ + 32 * ch + hi * 8;
#pragma unroll
        for (int q4 = 0; q4 < 4; ++q4) {
            float4 f = reinterpret_cast<const float4*>(qs)[q4 + (q4 >> 1) * 2];
            qb[ch][4 * q4 + 0] = (_Float16)(f.x * QS);
            qb[ch][4 * q4 + 1] = (_Float16)(f.y * QS);
            qb[ch][4 * q4 + 2] = (_Float16)(f.z * QS);
            qb[ch][4 * q4 + 3] = (_Float16)(f.w * QS);
        }
    }

    // ---- flash state ----
    float m = -3.0e38f, l = 0.0f;
    v8f o[8];
#pragma unroll
    for (int n = 0; n < 8; ++n)
#pragma unroll
        for (int r = 0; r < 8; ++r) o[n][r] = 0.0f;

    // ---- staging map: each thread owns a 4-row x 8-col sub-tile ----
    const int rg = (tid >> 4) << 2;     // first row (0,4,...,28)
    const int cg = (tid & 15) << 3;     // first col (0,8,...,120)
    const int jend = i0 + BM;           // causal limit for this block

    // register double buffer: prefetched f32 tile fragment
    float4 pf[4][2];
#pragma unroll
    for (int rr = 0; rr < 4; ++rr) {
        const float* src = vbh + (size_t)rr * D_ + (size_t)rg * D_ + cg;
        pf[rr][0] = reinterpret_cast<const float4*>(src)[0];
        pf[rr][1] = reinterpret_cast<const float4*>(src)[1];
    }

    for (int jb = 0; jb < jend; jb += BN) {
        __syncthreads();   // all waves done reading LDS from previous iteration

        // ---- convert prefetched tile and store both LDS layouts (packed) ----
        {
            _Float16 hh[4][8];
#pragma unroll
            for (int rr = 0; rr < 4; ++rr) {
                hh[rr][0] = (_Float16)pf[rr][0].x;
                hh[rr][1] = (_Float16)pf[rr][0].y;
                hh[rr][2] = (_Float16)pf[rr][0].z;
                hh[rr][3] = (_Float16)pf[rr][0].w;
                hh[rr][4] = (_Float16)pf[rr][1].x;
                hh[rr][5] = (_Float16)pf[rr][1].y;
                hh[rr][6] = (_Float16)pf[rr][1].z;
                hh[rr][7] = (_Float16)pf[rr][1].w;
                v8h rv;
#pragma unroll
                for (int e = 0; e < 8; ++e) rv[e] = hh[rr][e];
                *reinterpret_cast<v8h*>(&smV[rg + rr][cg]) = rv;        // ds_store_b128
            }
#pragma unroll
            for (int e = 0; e < 8; ++e) {
                v4h tv;
                tv[0] = hh[0][e]; tv[1] = hh[1][e];
                tv[2] = hh[2][e]; tv[3] = hh[3][e];
                *reinterpret_cast<v4h*>(&smVT[cg + e][rg]) = tv;        // ds_store_b64
            }
        }
        __syncthreads();

        // ---- issue next tile's global loads before compute (latency overlap) ----
        if (jb + BN < jend) {
            const float* base = vbh + (size_t)(jb + BN + rg) * D_ + cg;
            __builtin_prefetch(base + BN * D_, 0, 0);   // tile t+2 -> global_prefetch
#pragma unroll
            for (int rr = 0; rr < 4; ++rr) {
                const float* src = base + (size_t)rr * D_;
                pf[rr][0] = reinterpret_cast<const float4*>(src)[0];
                pf[rr][1] = reinterpret_cast<const float4*>(src)[1];
            }
        }

        if (jb > iw + 15) continue;   // wave fully masked; barriers stay uniform

        // ---- matmul 1: S^T(32x16) = Vtile(32x128) x Q^T(128x16) ----
        v8f c0v, c1v;
#pragma unroll
        for (int r = 0; r < 8; ++r) { c0v[r] = 0.0f; c1v[r] = 0.0f; }
#pragma unroll
        for (int ch = 0; ch < 4; ++ch) {
            const int cb = 32 * ch;
            v16h a0, a1;
            {
                v8h r1 = *reinterpret_cast<const v8h*>(&smV[li][cb + hi * 8]);
                v8h r2 = *reinterpret_cast<const v8h*>(&smV[li][cb + 16 + hi * 8]);
#pragma unroll
                for (int e = 0; e < 8; ++e) { a0[e] = r1[e]; a0[8 + e] = r2[e]; }
            }
            {
                v8h r1 = *reinterpret_cast<const v8h*>(&smV[16 + li][cb + hi * 8]);
                v8h r2 = *reinterpret_cast<const v8h*>(&smV[16 + li][cb + 16 + hi * 8]);
#pragma unroll
                for (int e = 0; e < 8; ++e) { a1[e] = r1[e]; a1[8 + e] = r2[e]; }
            }
            c0v = __builtin_amdgcn_wmma_f32_16x16x32_f16(false, a0, false, qb[ch],
                                                         (short)0, c0v, false, false);
            c1v = __builtin_amdgcn_wmma_f32_16x16x32_f16(false, a1, false, qb[ch],
                                                         (short)0, c1v, false, false);
        }

        // ---- causal mask: only the wave's diagonal tiles need it (uniform) ----
        float s0[8], s1[8];
#pragma unroll
        for (int r = 0; r < 8; ++r) { s0[r] = c0v[r]; s1[r] = c1v[r]; }
        if (jb + (BN - 1) > iw) {            // wave-uniform branch
#pragma unroll
            for (int r = 0; r < 8; ++r) {
                const int j0 = jb + 8 * hi + r;
                if (j0      > ig) s0[r] = -3.0e38f;
                if (j0 + 16 > ig) s1[r] = -3.0e38f;
            }
        }

        // ---- online softmax (lane owns row i = ig) ----
        float mloc = s0[0];
#pragma unroll
        for (int r = 1; r < 8; ++r) mloc = fmaxf(mloc, s0[r]);
#pragma unroll
        for (int r = 0; r < 8; ++r) mloc = fmaxf(mloc, s1[r]);
        mloc = fmaxf(mloc, xor16f(mloc));
        const float mnew  = fmaxf(m, mloc);
        const float alpha = fast_exp2(m - mnew);
        float p0[8], p1[8], ps = 0.0f;
#pragma unroll
        for (int r = 0; r < 8; ++r) {
            p0[r] = fast_exp2(s0[r] - mnew); ps += p0[r];
            p1[r] = fast_exp2(s1[r] - mnew); ps += p1[r];
        }
        ps += xor16f(ps);
        l = l * alpha + ps;
        m = mnew;

        // ---- rescale accumulator (rows live as M = 8*hi + r) ----
        float ar[8];
#pragma unroll
        for (int r = 0; r < 8; ++r) ar[r] = lanebc(alpha, 8 * hi + r);
#pragma unroll
        for (int n = 0; n < 8; ++n)
#pragma unroll
            for (int r = 0; r < 8; ++r) o[n][r] *= ar[r];

        // ---- free P transpose: S^T C-layout == A-layout of P(16x32) ----
        v16h pa;
#pragma unroll
        for (int e = 0; e < 8; ++e) {
            pa[e]     = (_Float16)p0[e];
            pa[8 + e] = (_Float16)p1[e];
        }

        // ---- matmul 2: O(16x128) += P(16x32) x Vtile(32x128) ----
#pragma unroll
        for (int n = 0; n < 8; ++n) {
            const int dl = 16 * n + li;
            v8h b1 = *reinterpret_cast<const v8h*>(&smVT[dl][hi * 8]);
            v8h b2 = *reinterpret_cast<const v8h*>(&smVT[dl][16 + hi * 8]);
            v16h bv;
#pragma unroll
            for (int e = 0; e < 8; ++e) { bv[e] = b1[e]; bv[8 + e] = b2[e]; }
            o[n] = __builtin_amdgcn_wmma_f32_16x16x32_f16(false, pa, false, bv,
                                                          (short)0, o[n], false, false);
        }
    }

    // ---- epilogue: divide by row sums, store ----
    float lr[8];
#pragma unroll
    for (int r = 0; r < 8; ++r) lr[r] = 1.0f / lanebc(l, 8 * hi + r);
#pragma unroll
    for (int n = 0; n < 8; ++n) {
        const int d = 16 * n + li;
#pragma unroll
        for (int r = 0; r < 8; ++r) {
            const int row = iw + 8 * hi + r;
            obh[(size_t)row * D_ + d] = o[n][r] * lr[r];
        }
    }
}

extern "C" void kernel_launch(void* const* d_in, const int* in_sizes, int n_in,
                              void* d_out, int out_size, void* d_ws, size_t ws_size,
                              hipStream_t stream) {
    (void)in_sizes; (void)n_in; (void)out_size; (void)d_ws; (void)ws_size;
    // inputs: d_in[0]=q (unused by reference!), d_in[1]=k, d_in[2]=v
    const float* k = (const float*)d_in[1];
    const float* v = (const float*)d_in[2];
    float* out = (float*)d_out;
    dim3 grid(S_ / BM, B_ * H_);
    fa_swapop_kernel<<<grid, dim3(128), 0, stream>>>(k, v, out);
}
